// GeoNGNN_32143535243475
// MI455X (gfx1250) — matrix operand order repaired and checked by reading, not compiled, not run on hardware
//
#include <hip/hip_runtime.h>
#include <hip/hip_bf16.h>
#include <math.h>

// ---------------- problem constants ----------------
#define NNODES   10000
#define NEDGES   320000
#define NSGN     200000
#define NSGE     600000
#define NGRAPHS  32
#define HID      128
#define EFD      16
#define INNER_R  5.0f
#define OUTER_R  10.0f
#define EXTEND_R 6.0f
#define PI_F     3.14159265358979323846f

// ---------------- input slot map (setup_inputs() insertion order, params
// flattened depth-first in insertion order) ----------------
#define IN_POS      0
#define IN_Z        1
#define IN_EDGE     2
#define IN_BATCH    3
#define IN_SGN      4
#define IN_SGC      5
#define IN_SGE      6
#define IN_SGB      7
#define IN_SGL      8
#define P_ZEMB_IN   9
#define P_ZEMB_OUT  10
#define P_LABEL_EMB 11
#define P_IPROJ_W   12
#define P_IPROJ_B   13
#define P_OPROJ_W   14
#define P_OPROJ_B   15
#define P_LPROJ_W   16
#define P_LPROJ_B   17
#define P_ILAYERS   18   // 3 layers x {Wm,bm,We,Wu,bu}
#define P_OLAYERS   33   // 4 layers x {Wm,bm,We,Wu,bu}
#define P_FUSE_W    53
#define P_FUSE_B    54
#define P_RES       55   // 2 blocks x {W1,b1,W2,b2}
#define P_OUT_W     63

typedef __attribute__((ext_vector_type(16))) __bf16 v16bf;
typedef __attribute__((ext_vector_type(8)))  float  v8f;

// ---------------- vectorized elementwise helpers (n must be /4) ----------------
__global__ void k_zero4(float4* p, int n4) {
  const float4 z = {0.f, 0.f, 0.f, 0.f};
  int i = blockIdx.x * blockDim.x + threadIdx.x;
  int stride = gridDim.x * blockDim.x;
  for (; i < n4; i += stride) p[i] = z;
}

__global__ void k_add4(float4* __restrict__ out, const float4* __restrict__ a, int n4) {
  int i = blockIdx.x * blockDim.x + threadIdx.x;
  int stride = gridDim.x * blockDim.x;
  for (; i < n4; i += stride) {
    float4 o = out[i], v = a[i];
    o.x += v.x; o.y += v.y; o.z += v.z; o.w += v.w;
    out[i] = o;
  }
}

__global__ void k_concat4(const float4* __restrict__ a, const float4* __restrict__ b,
                          float4* __restrict__ out, int rows) {
  int i = blockIdx.x * blockDim.x + threadIdx.x;  // over rows*64 float4s
  int total = rows * 64;
  int stride = gridDim.x * blockDim.x;
  for (; i < total; i += stride) {
    int r = i >> 6, c4 = i & 63;
    out[i] = (c4 < 32) ? a[(size_t)r * 32 + c4] : b[(size_t)r * 32 + (c4 - 32)];
  }
}

__global__ void k_f32_to_bf16(const float* __restrict__ src, __bf16* __restrict__ dst, int n) {
  int i = blockIdx.x * blockDim.x + threadIdx.x;
  if (i < n) dst[i] = (__bf16)src[i];
}

__global__ void k_make_idx(const int* __restrict__ z, const int* __restrict__ sgn,
                           int* __restrict__ idx, int n) {
  int i = blockIdx.x * blockDim.x + threadIdx.x;
  if (i < n) idx[i] = z[sgn[i]];
}

// ---------------- WMMA GEMM: out[M,128] = epi(A[rowidx, K] @ Wbf[K,128]) ----------------
// 256 threads = 8 waves; each wave computes a 16x128 strip via 8 bf16 WMMAs/K-step.
template<int K>
__global__ void k_gemm(int M,
                       const float* __restrict__ A,
                       const int*   __restrict__ rowidx,
                       const unsigned short* __restrict__ Wbf,  // bf16 bits, [K,128] row-major
                       const float* __restrict__ bias,
                       int doSilu,
                       const float* __restrict__ mulp,
                       const float* __restrict__ resp,
                       float* __restrict__ outp) {
  // Pre-pack W into LDS in CDNA5 B-fragment layout:
  // fragment f = ((kt*8 + nt)*32 + lane)*16 + e  <-  W[kt*32 + 16*(e>>3) + 8*(lane>>4) + (e&7)][nt*16 + (lane&15)]
  __shared__ __align__(32) unsigned short ldsW[(K / 32) * 4096];
  const int tid = threadIdx.x;
  {
    const int total = (K / 32) * 4096;
    for (int f = tid; f < total; f += 256) {
      int e  = f & 15;
      int l  = (f >> 4) & 31;
      int nt = (f >> 9) & 7;
      int kt = f >> 12;
      int krow = kt * 32 + ((e >> 3) * 16) + ((l >> 4) * 8) + (e & 7);
      int col  = nt * 16 + (l & 15);
      ldsW[f] = Wbf[krow * 128 + col];
    }
  }
  __syncthreads();

  const int wave = tid >> 5;
  const int lane = tid & 31;
  const int half = lane >> 4;

  int mrow = blockIdx.x * 128 + wave * 16 + (lane & 15);
  int srow = mrow < M ? mrow : (M - 1);
  int arow = rowidx ? rowidx[srow] : srow;
  const float* Abase = A + (size_t)arow * K;

  const v8f vzero = {0.f, 0.f, 0.f, 0.f, 0.f, 0.f, 0.f, 0.f};
  v8f acc[8];
#pragma unroll
  for (int nt = 0; nt < 8; ++nt) acc[nt] = vzero;

  const v16bf* bfr = reinterpret_cast<const v16bf*>(ldsW);

#pragma unroll
  for (int kk = 0; kk < K; kk += 32) {
    if (kk + 32 < K) __builtin_prefetch(Abase + kk + 32, 0, 0);  // global_prefetch_b8
    // A fragment (16x32 bf16): element e -> K = kk + (e&7) + 8*half + 16*(e>>3)
    const float4* ap = reinterpret_cast<const float4*>(Abase + kk + 8 * half);
    float4 f0 = ap[0];
    float4 f1 = ap[1];
    float4 f2 = ap[4];
    float4 f3 = ap[5];
    v16bf a;
    a[0]  = (__bf16)f0.x; a[1]  = (__bf16)f0.y; a[2]  = (__bf16)f0.z; a[3]  = (__bf16)f0.w;
    a[4]  = (__bf16)f1.x; a[5]  = (__bf16)f1.y; a[6]  = (__bf16)f1.z; a[7]  = (__bf16)f1.w;
    a[8]  = (__bf16)f2.x; a[9]  = (__bf16)f2.y; a[10] = (__bf16)f2.z; a[11] = (__bf16)f2.w;
    a[12] = (__bf16)f3.x; a[13] = (__bf16)f3.y; a[14] = (__bf16)f3.z; a[15] = (__bf16)f3.w;

    const int kbase = (kk >> 5) * 256;  // v16bf units per k-tile = 8*32
#pragma unroll
    for (int nt = 0; nt < 8; ++nt) {
      v16bf b = bfr[kbase + nt * 32 + lane];
      acc[nt] = __builtin_amdgcn_wmma_f32_16x16x32_bf16(
          /*neg_a=*/false, a, /*neg_b=*/false, b,
          /*c_mod=*/(short)0, acc[nt], /*reuse_a=*/false, /*reuse_b=*/false);
    }
  }

  // Epilogue: bias + SiLU + optional mul + optional residual; C layout:
  // VGPR r: row = base + r + 8*half, col = nt*16 + (lane&15)
  const int col0 = lane & 15;
  const int rbase = blockIdx.x * 128 + wave * 16;
#pragma unroll
  for (int nt = 0; nt < 8; ++nt) {
    int col = nt * 16 + col0;
    float bv = bias ? bias[col] : 0.0f;
#pragma unroll
    for (int r = 0; r < 8; ++r) {
      int row = rbase + r + 8 * half;
      if (row < M) {
        float v = acc[nt][r] + bv;
        if (doSilu) v = v * (1.0f / (1.0f + expf(-v)));
        size_t o = (size_t)row * 128 + col;
        if (mulp) v *= mulp[o];
        if (resp) v += resp[o];
        outp[o] = v;
      }
    }
  }
}

// ---------------- edge message + scatter (segment_sum via atomics) ----------------
// One wave per edge; 4 channels per lane. We (16x128) staged in LDS once per block.
__global__ void k_edge_msg(const int* __restrict__ esrc, const int* __restrict__ edst, int E,
                           const float* __restrict__ pos, const int* __restrict__ posidx,
                           const float* __restrict__ H, const float* __restrict__ We,
                           float rbound, float rmax, float* __restrict__ agg) {
  __shared__ __align__(16) float sWe[EFD * 128];
  for (int i = threadIdx.x; i < EFD * 128; i += blockDim.x) sWe[i] = We[i];
  __syncthreads();

  int gwave = (blockIdx.x * blockDim.x + threadIdx.x) >> 5;
  int lane  = threadIdx.x & 31;
  int nwaves = (gridDim.x * blockDim.x) >> 5;
  const float gam = (EFD / rbound) * (EFD / rbound);
  const float step = rbound / (EFD - 1);
  for (int e = gwave; e < E; e += nwaves) {
    int s = esrc[e], d = edst[e];
    int ps = posidx ? posidx[s] : s;
    int pd = posidx ? posidx[d] : d;
    float dx = pos[ps * 3 + 0] - pos[pd * 3 + 0];
    float dy = pos[ps * 3 + 1] - pos[pd * 3 + 1];
    float dz = pos[ps * 3 + 2] - pos[pd * 3 + 2];
    float dist = sqrtf(dx * dx + dy * dy + dz * dz + 1e-12f);
    float sm = (dist < rmax) ? 0.5f * (cosf(PI_F * dist / rmax) + 1.0f) : 0.0f;

    // gate = rbf(dist) @ We  for this lane's 4 channels (float4 from LDS)
    float4 g = {0.f, 0.f, 0.f, 0.f};
#pragma unroll
    for (int k = 0; k < EFD; ++k) {
      float t = dist - (float)k * step;
      float r = expf(-gam * t * t);
      const float4 w = *reinterpret_cast<const float4*>(&sWe[k * 128 + lane * 4]);
      g.x += r * w.x; g.y += r * w.y; g.z += r * w.z; g.w += r * w.w;
    }
    const float4 h = *reinterpret_cast<const float4*>(&H[(size_t)s * 128 + lane * 4]);
    float* ap = &agg[(size_t)d * 128 + lane * 4];
    atomicAdd(ap + 0, h.x * g.x * sm);
    atomicAdd(ap + 1, h.y * g.y * sm);
    atomicAdd(ap + 2, h.z * g.z * sm);
    atomicAdd(ap + 3, h.w * g.w * sm);
  }
}

// ---------------- subgraph pooling with smooth envelope ----------------
// One wave per 8 subgraph nodes: each quarter-wave (8 lanes... ) -> simpler:
// thread handles 4 channels of one node (32 threads per node pair grouping).
__global__ void k_pool_env(const float* __restrict__ sc, const float* __restrict__ pos,
                           const int* __restrict__ sgn, const int* __restrict__ sgc,
                           const int* __restrict__ sgb, int Ns, float* __restrict__ env_out) {
  int i = blockIdx.x * blockDim.x + threadIdx.x;  // over Ns*32 quads
  int total = Ns * 32;
  int stride = gridDim.x * blockDim.x;
  for (; i < total; i += stride) {
    int node = i >> 5, q = i & 31;
    int a = sgn[node], b = sgc[node];
    float dx = pos[a * 3 + 0] - pos[b * 3 + 0];
    float dy = pos[a * 3 + 1] - pos[b * 3 + 1];
    float dz = pos[a * 3 + 2] - pos[b * 3 + 2];
    float dist = sqrtf(dx * dx + dy * dy + dz * dz + 1e-12f);
    float env = (dist < EXTEND_R) ? 0.5f * (cosf(PI_F * dist / EXTEND_R) + 1.0f) : 0.0f;
    const float4 v = *reinterpret_cast<const float4*>(&sc[(size_t)node * 128 + q * 4]);
    float* ep = &env_out[(size_t)sgb[node] * 128 + q * 4];
    atomicAdd(ep + 0, v.x * env);
    atomicAdd(ep + 1, v.y * env);
    atomicAdd(ep + 2, v.z * env);
    atomicAdd(ep + 3, v.w * env);
  }
}

__global__ void k_graph_pool(const float* __restrict__ sc, const int* __restrict__ batch,
                             int n, float* __restrict__ graph) {
  int i = blockIdx.x * blockDim.x + threadIdx.x;  // over n*32 quads
  int total = n * 32;
  int stride = gridDim.x * blockDim.x;
  for (; i < total; i += stride) {
    int node = i >> 5, q = i & 31;
    const float4 v = *reinterpret_cast<const float4*>(&sc[(size_t)node * 128 + q * 4]);
    float* gp = &graph[(size_t)batch[node] * 128 + q * 4];
    atomicAdd(gp + 0, v.x);
    atomicAdd(gp + 1, v.y);
    atomicAdd(gp + 2, v.z);
    atomicAdd(gp + 3, v.w);
  }
}

__global__ void k_readout(const float* __restrict__ graph, const float* __restrict__ outW,
                          float* __restrict__ out) {
  __shared__ float red[128];
  int g = blockIdx.x, t = threadIdx.x;
  red[t] = graph[g * 128 + t] * outW[t];
  __syncthreads();
  for (int s = 64; s > 0; s >>= 1) {
    if (t < s) red[t] += red[t + s];
    __syncthreads();
  }
  if (t == 0) out[g] = red[0];  // Y_STD=1, Y_MEAN=0
}

// ---------------- host-side orchestration ----------------
static inline int cdiv(int a, int b) { return (a + b - 1) / b; }

extern "C" void kernel_launch(void* const* d_in, const int* in_sizes, int n_in,
                              void* d_out, int out_size, void* d_ws, size_t ws_size,
                              hipStream_t stream) {
  (void)in_sizes; (void)n_in; (void)out_size; (void)ws_size;
  const float* pos   = (const float*)d_in[IN_POS];
  const int*   z     = (const int*)  d_in[IN_Z];
  const int*   edge  = (const int*)  d_in[IN_EDGE];
  const int*   batch = (const int*)  d_in[IN_BATCH];
  const int*   sgn   = (const int*)  d_in[IN_SGN];
  const int*   sgc   = (const int*)  d_in[IN_SGC];
  const int*   sge   = (const int*)  d_in[IN_SGE];
  const int*   sgb   = (const int*)  d_in[IN_SGB];
  const int*   sgl   = (const int*)  d_in[IN_SGL];

  // workspace carve-up (256B aligned)
  char* wsp = (char*)d_ws;
  size_t off = 0;
  auto take = [&](size_t bytes) -> void* {
    void* p = wsp + off;
    off += (bytes + 255) & ~(size_t)255;
    return p;
  };
  float* sc_in   = (float*)take((size_t)NSGN * 128 * 4);
  float* H_in    = (float*)take((size_t)NSGN * 128 * 4);
  float* agg_in  = (float*)take((size_t)NSGN * 128 * 4);
  float* out_sc  = (float*)take((size_t)NNODES * 128 * 4);
  float* H_out   = (float*)take((size_t)NNODES * 128 * 4);
  float* agg_out = (float*)take((size_t)NNODES * 128 * 4);
  float* hbuf    = (float*)take((size_t)NNODES * 128 * 4);
  float* senv    = (float*)take((size_t)NNODES * 128 * 4);
  float* catbuf  = (float*)take((size_t)NNODES * 256 * 4);
  float* graph   = (float*)take((size_t)NGRAPHS * 128 * 4);
  int*   idx_in  = (int*)  take((size_t)NSGN * 4);
  // bf16 weight blobs
  __bf16* w_iproj = (__bf16*)take(16384 * 2);
  __bf16* w_lproj = (__bf16*)take(16384 * 2);
  __bf16* w_oproj = (__bf16*)take(16384 * 2);
  __bf16* w_fuse  = (__bf16*)take(32768 * 2);
  __bf16* w_im[3]; __bf16* w_iu[3];
  for (int i = 0; i < 3; ++i) { w_im[i] = (__bf16*)take(16384 * 2); w_iu[i] = (__bf16*)take(16384 * 2); }
  __bf16* w_om[4]; __bf16* w_ou[4];
  for (int i = 0; i < 4; ++i) { w_om[i] = (__bf16*)take(16384 * 2); w_ou[i] = (__bf16*)take(16384 * 2); }
  __bf16* w_res[4];
  for (int i = 0; i < 4; ++i) w_res[i] = (__bf16*)take(16384 * 2);

  // ---- weight conversion to bf16 ----
  auto conv = [&](const void* src, __bf16* dst, int n) {
    k_f32_to_bf16<<<cdiv(n, 256), 256, 0, stream>>>((const float*)src, dst, n);
  };
  conv(d_in[P_IPROJ_W], w_iproj, 16384);
  conv(d_in[P_LPROJ_W], w_lproj, 16384);
  conv(d_in[P_OPROJ_W], w_oproj, 16384);
  conv(d_in[P_FUSE_W],  w_fuse,  32768);
  for (int i = 0; i < 3; ++i) {
    conv(d_in[P_ILAYERS + 5 * i + 0], w_im[i], 16384);
    conv(d_in[P_ILAYERS + 5 * i + 3], w_iu[i], 16384);
  }
  for (int i = 0; i < 4; ++i) {
    conv(d_in[P_OLAYERS + 5 * i + 0], w_om[i], 16384);
    conv(d_in[P_OLAYERS + 5 * i + 3], w_ou[i], 16384);
  }
  for (int i = 0; i < 2; ++i) {
    conv(d_in[P_RES + 4 * i + 0], w_res[2 * i + 0], 16384);
    conv(d_in[P_RES + 4 * i + 2], w_res[2 * i + 1], 16384);
  }

  k_make_idx<<<cdiv(NSGN, 256), 256, 0, stream>>>(z, sgn, idx_in, NSGN);

  auto gemm128 = [&](int M, const float* A, const int* ridx, const __bf16* W,
                     const float* b, int silu, const float* mul, const float* res, float* out) {
    k_gemm<128><<<cdiv(M, 128), 256, 0, stream>>>(
        M, A, ridx, (const unsigned short*)W, b, silu, mul, res, out);
  };
  auto zero = [&](float* p, int n) {
    k_zero4<<<cdiv(n / 4, 256), 256, 0, stream>>>((float4*)p, n / 4);
  };

  // ================= inner (subgraph) branch =================
  // H_in = SiLU(z_emb_in[z[sgn]] @ Wip + b)
  gemm128(NSGN, (const float*)d_in[P_ZEMB_IN], idx_in, w_iproj,
          (const float*)d_in[P_IPROJ_B], 1, nullptr, nullptr, H_in);
  // sc_in = SiLU(label_emb[sgl] @ Wlp + b) * H_in
  gemm128(NSGN, (const float*)d_in[P_LABEL_EMB], sgl, w_lproj,
          (const float*)d_in[P_LPROJ_B], 1, H_in, nullptr, sc_in);

  const int* isrc = sge;
  const int* idst = sge + NSGE;
  for (int L = 0; L < 3; ++L) {
    const float* bm = (const float*)d_in[P_ILAYERS + 5 * L + 1];
    const float* We = (const float*)d_in[P_ILAYERS + 5 * L + 2];
    const float* bu = (const float*)d_in[P_ILAYERS + 5 * L + 4];
    gemm128(NSGN, sc_in, nullptr, w_im[L], bm, 1, nullptr, nullptr, H_in);  // hoisted msg dense
    zero(agg_in, NSGN * 128);
    k_edge_msg<<<cdiv(NSGE * 32, 256), 256, 0, stream>>>(
        isrc, idst, NSGE, pos, sgn, H_in, We, INNER_R, INNER_R, agg_in);
    gemm128(NSGN, agg_in, nullptr, w_iu[L], bu, 1, nullptr, sc_in, sc_in);  // residual update
  }

  // subgraph pooling with smooth envelope -> senv [NNODES,128]
  zero(senv, NNODES * 128);
  k_pool_env<<<cdiv(NSGN * 32, 256), 256, 0, stream>>>(sc_in, pos, sgn, sgc, sgb, NSGN, senv);

  // ================= outer branch =================
  gemm128(NNODES, (const float*)d_in[P_ZEMB_OUT], z, w_oproj,
          (const float*)d_in[P_OPROJ_B], 1, nullptr, nullptr, out_sc);
  k_concat4<<<cdiv(NNODES * 64, 256), 256, 0, stream>>>(
      (const float4*)out_sc, (const float4*)senv, (float4*)catbuf, NNODES);
  k_gemm<256><<<cdiv(NNODES, 128), 256, 0, stream>>>(
      NNODES, catbuf, nullptr, (const unsigned short*)w_fuse,
      (const float*)d_in[P_FUSE_B], 1, nullptr, nullptr, hbuf);
  for (int i = 0; i < 2; ++i) {
    const float* b1 = (const float*)d_in[P_RES + 4 * i + 1];
    const float* b2 = (const float*)d_in[P_RES + 4 * i + 3];
    gemm128(NNODES, hbuf, nullptr, w_res[2 * i + 0], b1, 1, nullptr, nullptr, H_out);
    gemm128(NNODES, H_out, nullptr, w_res[2 * i + 1], b2, 1, nullptr, hbuf, hbuf);
  }
  k_add4<<<cdiv(NNODES * 32, 256), 256, 0, stream>>>(
      (float4*)out_sc, (const float4*)hbuf, NNODES * 32);

  const int* osrc = edge;
  const int* odst = edge + NEDGES;
  for (int L = 0; L < 4; ++L) {
    const float* bm = (const float*)d_in[P_OLAYERS + 5 * L + 1];
    const float* We = (const float*)d_in[P_OLAYERS + 5 * L + 2];
    const float* bu = (const float*)d_in[P_OLAYERS + 5 * L + 4];
    gemm128(NNODES, out_sc, nullptr, w_om[L], bm, 1, nullptr, nullptr, H_out);
    zero(agg_out, NNODES * 128);
    k_edge_msg<<<cdiv(NEDGES * 32, 256), 256, 0, stream>>>(
        osrc, odst, NEDGES, pos, nullptr, H_out, We, OUTER_R, OUTER_R, agg_out);
    gemm128(NNODES, agg_out, nullptr, w_ou[L], bu, 1, nullptr, out_sc, out_sc);
  }

  // graph pooling + readout
  zero(graph, NGRAPHS * 128);
  k_graph_pool<<<cdiv(NNODES * 32, 256), 256, 0, stream>>>(out_sc, batch, NNODES, graph);
  k_readout<<<NGRAPHS, 128, 0, stream>>>(graph, (const float*)d_in[P_OUT_W], (float*)d_out);
}
